// Net2d_39865886441872
// MI455X (gfx1250) — compile-verified, matrix operand order checked by reading, not run on hardware
//
#include <hip/hip_runtime.h>

// ---------------------------------------------------------------------------
// FNO2D on gfx1250 (MI455X).  All heavy math is batched GEMMs on
// v_wmma_f32_16x16x32_bf16 (wave32).  FFTs are mode-truncated dense DFT
// matrices (only 40x20 modes survive).  Shapes are padded so fragment
// loads/stores need no bounds checks; each bf16 fragment is exactly two
// global_load_b128's.  B operands are stored transposed (n-major) so the
// A-style fragment loader serves both operands.  K-loops are software-
// pipelined (register double-buffering) so b128 loads overlap the MMAs.
// Complex GEMM pairs and the pointwise+inverse-W pair are fused as
// dual-operand GEMMs (two K-loops, one store, no C read-modify-write);
// the fused pointwise/inverse-W kernel reduces BN sum/sumsq in its
// epilogue.  All layout transposes go through 64x64 LDS tiles so both
// global sides are coalesced 128B transactions.
// ---------------------------------------------------------------------------

typedef __bf16  bf16x16 __attribute__((ext_vector_type(16)));
typedef __bf16  bf16x8  __attribute__((ext_vector_type(8)));
typedef float   f32x8   __attribute__((ext_vector_type(8)));

#define NROWS (8L*256*256)

// --------- fragment loader: 16-bit 16x32 layout = two 16B runs -------------
__device__ __forceinline__ bf16x16 load_frag(const __bf16* __restrict__ P,
                                             long rowbase, int k0, int hi) {
  union { bf16x16 v; bf16x8 q[2]; } u;
  const __bf16* p = P + rowbase + k0 + hi * 8;
  u.q[0] = *(const bf16x8*)(p);
  u.q[1] = *(const bf16x8*)(p + 16);
  return u.v;
}

#define WMMA_BF16(a, b, c) \
  __builtin_amdgcn_wmma_f32_16x16x32_bf16(false, (a), false, (b), (short)0, (c), false, false)

// Software-pipelined K-loop: 16x64 strip of accumulators, 4 WMMAs per chunk.
__device__ __forceinline__ void mma_loop(const __bf16* __restrict__ A,
                                         const __bf16* __restrict__ Bt,
                                         long aRow, long bRow, long bT, int K, int hi,
                                         f32x8& a0, f32x8& a1, f32x8& a2, f32x8& a3) {
  bf16x16 ac = load_frag(A,  aRow,          0, hi);
  bf16x16 c0 = load_frag(Bt, bRow,          0, hi);
  bf16x16 c1 = load_frag(Bt, bRow + bT,     0, hi);
  bf16x16 c2 = load_frag(Bt, bRow + 2 * bT, 0, hi);
  bf16x16 c3 = load_frag(Bt, bRow + 3 * bT, 0, hi);
  for (int k0 = 32; k0 < K; k0 += 32) {
    bf16x16 an = load_frag(A,  aRow,          k0, hi);
    bf16x16 n0 = load_frag(Bt, bRow,          k0, hi);
    bf16x16 n1 = load_frag(Bt, bRow + bT,     k0, hi);
    bf16x16 n2 = load_frag(Bt, bRow + 2 * bT, k0, hi);
    bf16x16 n3 = load_frag(Bt, bRow + 3 * bT, k0, hi);
    a0 = WMMA_BF16(ac, c0, a0);
    a1 = WMMA_BF16(ac, c1, a1);
    a2 = WMMA_BF16(ac, c2, a2);
    a3 = WMMA_BF16(ac, c3, a3);
    ac = an; c0 = n0; c1 = n1; c2 = n2; c3 = n3;
  }
  a0 = WMMA_BF16(ac, c0, a0);
  a1 = WMMA_BF16(ac, c1, a1);
  a2 = WMMA_BF16(ac, c2, a2);
  a3 = WMMA_BF16(ac, c3, a3);
}

// ------------------------- single-operand batched GEMM ----------------------
__global__ __launch_bounds__(32)
void gemm_wmma(const __bf16* __restrict__ A, const __bf16* __restrict__ Bt,
               float* __restrict__ C,
               int M, int N, int K, long lda, long ldbt, long ldc,
               long baseA, long baseB, long baseC,
               long sA1, long sA2, long sB1, long sB2, long sC1, long sC2) {
  long b1 = blockIdx.z, b2 = blockIdx.y;
  int ngrp = N >> 6;
  int tm = (int)blockIdx.x / ngrp, ng = (int)blockIdx.x % ngrp;
  int row0 = tm << 4, col0 = ng << 6;
  int lane = threadIdx.x & 31, lr = lane & 15, hi = lane >> 4;

  long aRow = baseA + b1 * sA1 + b2 * sA2 + (long)(row0 + lr) * lda;
  long bRow = baseB + b1 * sB1 + b2 * sB2 + (long)(col0 + lr) * ldbt;

  f32x8 acc0, acc1, acc2, acc3;
#pragma unroll
  for (int r = 0; r < 8; ++r) { acc0[r] = 0.f; acc1[r] = 0.f; acc2[r] = 0.f; acc3[r] = 0.f; }

  mma_loop(A, Bt, aRow, bRow, (long)16 * ldbt, K, hi, acc0, acc1, acc2, acc3);

  long mrow = baseC + b1 * sC1 + b2 * sC2 + (long)(row0 + (hi << 3)) * ldc;
#pragma unroll
  for (int t = 0; t < 4; ++t) {
    f32x8 acc = (t == 0) ? acc0 : (t == 1) ? acc1 : (t == 2) ? acc2 : acc3;
    long off = mrow + col0 + (t << 4) + lr;
#pragma unroll
    for (int r = 0; r < 8; ++r) C[off + (long)r * ldc] = acc[r];
  }
}

// ------------------------- dual-operand batched GEMM ------------------------
// C = A1@Bt1^T + A2@Bt2^T (two pipelined K loops, one store).
// Optional: column-split store (spectral repack) and BN sum/sumsq epilogue.
__global__ __launch_bounds__(32)
void gemm_wmma_dual(const __bf16* __restrict__ A1, const __bf16* __restrict__ Bt1,
                    const __bf16* __restrict__ A2, const __bf16* __restrict__ Bt2,
                    float* __restrict__ C,
                    int M, int N, int K1, int K2,
                    long lda1, long ldbt1, long lda2, long ldbt2, long ldc,
                    long baseA1, long baseB1, long baseA2, long baseB2, long baseC,
                    long sA11, long sA12, long sB11, long sB12,
                    long sA21, long sA22, long sB21, long sB22,
                    long sC1, long sC2,
                    int nsplit, long splitadd, float* __restrict__ stats) {
  long b1 = blockIdx.z, b2 = blockIdx.y;
  int ngrp = N >> 6;
  int tm = (int)blockIdx.x / ngrp, ng = (int)blockIdx.x % ngrp;
  int row0 = tm << 4, col0 = ng << 6;
  int lane = threadIdx.x & 31, lr = lane & 15, hi = lane >> 4;

  long a1Row = baseA1 + b1 * sA11 + b2 * sA12 + (long)(row0 + lr) * lda1;
  long b1Row = baseB1 + b1 * sB11 + b2 * sB12 + (long)(col0 + lr) * ldbt1;
  long a2Row = baseA2 + b1 * sA21 + b2 * sA22 + (long)(row0 + lr) * lda2;
  long b2Row = baseB2 + b1 * sB21 + b2 * sB22 + (long)(col0 + lr) * ldbt2;

  f32x8 acc0, acc1, acc2, acc3;
#pragma unroll
  for (int r = 0; r < 8; ++r) { acc0[r] = 0.f; acc1[r] = 0.f; acc2[r] = 0.f; acc3[r] = 0.f; }

  mma_loop(A1, Bt1, a1Row, b1Row, (long)16 * ldbt1, K1, hi, acc0, acc1, acc2, acc3);
  mma_loop(A2, Bt2, a2Row, b2Row, (long)16 * ldbt2, K2, hi, acc0, acc1, acc2, acc3);

  long mrow = baseC + b1 * sC1 + b2 * sC2 + (long)(row0 + (hi << 3)) * ldc;
#pragma unroll
  for (int t = 0; t < 4; ++t) {
    f32x8 acc = (t == 0) ? acc0 : (t == 1) ? acc1 : (t == 2) ? acc2 : acc3;
    int n = col0 + (t << 4) + lr;
    long nadd = (nsplit > 0 && n >= nsplit) ? splitadd : 0;
    long off = mrow + n + nadd;
#pragma unroll
    for (int r = 0; r < 8; ++r) C[off + (long)r * ldc] = acc[r];
  }

  if (stats) {                              // BN partial sums (column == channel)
#pragma unroll
    for (int t = 0; t < 4; ++t) {
      f32x8 acc = (t == 0) ? acc0 : (t == 1) ? acc1 : (t == 2) ? acc2 : acc3;
      float s = 0.f, s2 = 0.f;
#pragma unroll
      for (int r = 0; r < 8; ++r) { s += acc[r]; s2 += acc[r] * acc[r]; }
      s  += __shfl_xor(s, 16, 32);          // fold the two row-halves of column n
      s2 += __shfl_xor(s2, 16, 32);
      if (hi == 0) {
        int n = col0 + (t << 4) + lr;
        atomicAdd(&stats[n], s);
        atomicAdd(&stats[64 + n], s2);
      }
    }
  }
}

// --------------------------- small prep kernels -----------------------------

// DFT tables, all scaled by 1/16 (ortho).  Padded regions are zero.
__global__ void build_dft_tables(__bf16* DW, __bf16* MH1, __bf16* MH2,
                                 __bf16* EH1, __bf16* EH2, __bf16* G) {
  int idx = blockIdx.x * blockDim.x + threadIdx.x;
  const float S = 1.0f / 16.0f;
  const float TP = 6.283185307179586f;
  if (idx < 48 * 256) {                    // fwd W: rows 0..19 cos, 20..39 -sin, 40..47 zero
    int r = idx / 256, w = idx % 256;
    float v = 0.f;
    if (r < 40) {
      int kw = (r < 20) ? r : r - 20;
      float th = TP * (float)w * (float)kw / 256.0f;
      v = ((r < 20) ? __cosf(th) : -__sinf(th)) * S;
    }
    DW[idx] = (__bf16)v;
  }
  if (idx < 80 * 256) {                    // fwd H stacked complex operators
    int r = idx / 256, h = idx % 256;
    int j = (r < 40) ? r : r - 40;
    int kh = (j < 20) ? j : 216 + j;       // 236..255
    float th = TP * (float)h * (float)kh / 256.0f;
    float c = __cosf(th) * S, s = __sinf(th) * S;
    MH1[idx] = (__bf16)((r < 40) ? c : -s);   // [Dr ; Di], Di=-s
    MH2[idx] = (__bf16)((r < 40) ? s :  c);   // [-Di ; Dr]
  }
  if (idx < 256 * 96) {                    // inv H stacked-K operators, cols 80..95 zero
    int h = idx / 96, col = idx % 96;
    float v1 = 0.f, v2 = 0.f;
    if (col < 80) {
      int j = (col < 40) ? col : col - 40;
      int kh = (j < 20) ? j : 216 + j;
      float th = TP * (float)h * (float)kh / 256.0f;
      float c = __cosf(th) * S, s = __sinf(th) * S;
      v1 = (col < 40) ? c : -s;            // Zr = [Er | -Ei] [Or;Oi]
      v2 = (col < 40) ? s :  c;            // Zi = [Ei |  Er] [Or;Oi]
    }
    EH1[idx] = (__bf16)v1;
    EH2[idx] = (__bf16)v2;
  }
  if (idx < 256 * 64) {                    // inv rFFT along W, cols 40..63 zero
    int w = idx / 64, col = idx % 64;
    float v = 0.f;
    if (col < 40) {
      int kw = (col < 20) ? col : col - 20;
      float th = TP * (float)w * (float)kw / 256.0f;
      v = (col < 20) ? ((kw == 0) ? 1.0f :  2.0f * __cosf(th)) * S
                     : ((kw == 0) ? 0.0f : -2.0f * __sinf(th)) * S;
    }
    G[idx] = (__bf16)v;
  }
}

// cw[l] is [o][i] == Bt layout for the pointwise GEMM: plain cast.
__global__ void prep_cwb(const float* __restrict__ cw, __bf16* __restrict__ cwB, int l) {
  int idx = blockIdx.x * blockDim.x + threadIdx.x;
  if (idx < 64 * 64) cwB[idx] = (__bf16)cw[(long)l * 4096 + idx];
}

// fc1_w [64][128] -> W1t [128][64] (n-major).
__global__ void prep_w1t(const float* __restrict__ w1, __bf16* __restrict__ W1t) {
  int idx = blockIdx.x * blockDim.x + threadIdx.x;
  if (idx >= 128 * 64) return;
  int k = idx & 63, n = idx >> 6;
  W1t[idx] = (__bf16)w1[k * 128 + n];
}

// Spectral weights -> transposed stacked matrices Wmt[(j,kw)][n(128)][ci(64)].
__global__ void prep_wm(const float* __restrict__ sw1, const float* __restrict__ sw2,
                        __bf16* __restrict__ Wm1, __bf16* __restrict__ Wm2, int l) {
  long idx = (long)blockIdx.x * blockDim.x + threadIdx.x;
  if (idx >= 40L * 20 * 128 * 64) return;
  int ci = (int)(idx & 63);
  int n  = (int)((idx >> 6) & 127);
  int rem = (int)(idx >> 13);
  int kw = rem % 20, j = rem / 20;
  int co = (n < 64) ? n : n - 64;
  const float* sw = (j < 20) ? sw1 : sw2;
  int x = (j < 20) ? j : j - 20;
  long base = (((((long)l * 64 + ci) * 64 + co) * 20 + x) * 20 + kw) * 2;
  float Wr = sw[base], Wi = sw[base + 1];
  Wm1[idx] = (__bf16)((n < 64) ? Wr : Wi);
  Wm2[idx] = (__bf16)((n < 64) ? -Wi : Wr);
}

// ---------------- LDS-tiled transposes (both sides coalesced) ---------------

// hB [bh][w][c] -> hBt [bh][c][w], 64x64 bf16 tiles.
__global__ __launch_bounds__(256)
void transpose_hbt(const __bf16* __restrict__ hB, __bf16* __restrict__ hBt) {
  __shared__ __bf16 tile[64][65];
  long bh = blockIdx.x >> 2;
  int w0 = ((int)blockIdx.x & 3) << 6;
  int tid = threadIdx.x;
  int c = tid & 63, q = tid >> 6;
  long base = bh * 16384;
#pragma unroll
  for (int p = 0; p < 16; ++p) {
    int w = q + p * 4;
    tile[w][c] = hB[base + (long)(w0 + w) * 64 + c];
  }
  __syncthreads();
#pragma unroll
  for (int p = 0; p < 16; ++p) {
    int cc = q + p * 4;
    hBt[base + (long)cc * 256 + w0 + c] = tile[c][cc];
  }
}

// Xf [b][h][48][64] f32 -> Xbt [b][40(r)][64(c)][256(h)] bf16, LDS tiles.
__global__ __launch_bounds__(256)
void cvt_xbt(const float* __restrict__ Xf, __bf16* __restrict__ Xbt) {
  __shared__ __bf16 tile[64][65];
  int blk = blockIdx.x;                    // 8*40*4
  int h0 = (blk & 3) << 6;
  int r  = (blk >> 2) % 40;
  int b  = (blk >> 2) / 40;
  int tid = threadIdx.x;
  int c = tid & 63, q = tid >> 6;
#pragma unroll
  for (int p = 0; p < 16; ++p) {
    int hh = q + p * 4;
    tile[hh][c] = (__bf16)Xf[(((long)b * 256 + h0 + hh) * 48 + r) * 64 + c];
  }
  __syncthreads();
  long obase = (((long)b * 40 + r) * 64) * 256;
#pragma unroll
  for (int p = 0; p < 16; ++p) {
    int cc = q + p * 4;
    Xbt[obase + (long)cc * 256 + h0 + c] = tile[c][cc];
  }
}

// Of [16][20][80][64] f32 -> Obt [8][20][64(c)][96(r)] bf16, rows>=80 zero.
// Small + L2-resident: simple form is fine.
__global__ void cvt_obt(const float* __restrict__ Of, __bf16* __restrict__ Obt) {
  long idx = (long)blockIdx.x * blockDim.x + threadIdx.x;
  if (idx >= 8L * 20 * 64 * 96) return;
  int r = (int)(idx % 96);
  long rest = idx / 96;
  int c = (int)(rest & 63);
  int v = (int)(rest >> 6);
  int kw = v % 20, b = v / 20;
  float val = (r < 80) ? Of[(long)(b * 20 + kw) * 5120 + r * 64 + c] : 0.0f;
  Obt[idx] = (__bf16)val;
}

// Zf [b][h][40][64] f32 -> Zbt [b][h][64(c)][64(r)] bf16, rows>=40 zero.
__global__ void cvt_zbt(const float* __restrict__ Zf, __bf16* __restrict__ Zbt) {
  long idx = (long)blockIdx.x * blockDim.x + threadIdx.x;
  if (idx >= 8L * 256 * 64 * 64) return;
  int r = (int)(idx & 63);
  int c = (int)((idx >> 6) & 63);
  long bh = idx >> 12;                     // b*256+h
  float val = (r < 40) ? Zf[(bh * 40 + r) * 64 + c] : 0.0f;
  Zbt[idx] = (__bf16)val;
}

__global__ void f32_to_bf16(const float* __restrict__ in, __bf16* __restrict__ out, long n) {
  long i = (long)blockIdx.x * blockDim.x + threadIdx.x;
  if (i < n) out[i] = (__bf16)in[i];
}

// fc0 lift (K=12, VALU): writes hB only; hBt produced by transpose_hbt.
__global__ void fc0_kernel(const float* __restrict__ x, const float* __restrict__ w,
                           const float* __restrict__ b, __bf16* __restrict__ hB) {
  long idx = (long)blockIdx.x * blockDim.x + threadIdx.x;
  if (idx >= NROWS * 64) return;
  int c = (int)(idx & 63);
  long n = idx >> 6;
  float s = b[c];
#pragma unroll
  for (int t = 0; t < 12; ++t) s += x[n * 12 + t] * w[t * 64 + c];
  hB[idx] = (__bf16)s;
}

// BN apply, LDS-tiled: writes hB coalesced and hBt coalesced via transpose.
__global__ __launch_bounds__(256)
void bn_apply(const float* __restrict__ x, const float* __restrict__ stats,
              const float* __restrict__ gamma, const float* __restrict__ beta,
              __bf16* __restrict__ hB, __bf16* __restrict__ hBt, int relu) {
  __shared__ __bf16 tile[64][65];
  long bh = blockIdx.x >> 2;
  int w0 = ((int)blockIdx.x & 3) << 6;
  int tid = threadIdx.x;
  int c = tid & 63, q = tid >> 6;
  long base = bh * 16384;
  const float invN = 1.0f / (float)NROWS;
  float mean = stats[c] * invN;
  float var  = stats[64 + c] * invN - mean * mean;
  float sc   = gamma[c] * rsqrtf(var + 1e-5f);
  float sh   = beta[c];
#pragma unroll
  for (int p = 0; p < 16; ++p) {
    int w = q + p * 4;
    long off = base + (long)(w0 + w) * 64 + c;
    float v = (x[off] - mean) * sc + sh;
    if (relu) v = fmaxf(v, 0.0f);
    __bf16 bv = (__bf16)v;
    hB[off] = bv;
    tile[w][c] = bv;
  }
  __syncthreads();
#pragma unroll
  for (int p = 0; p < 16; ++p) {
    int cc = q + p * 4;
    hBt[base + (long)cc * 256 + w0 + c] = tile[c][cc];
  }
}

// Fused head: out = relu(h @ W1 + b1) @ w2 + b2; 16 rows/wave, WMMA for fc1.
__global__ __launch_bounds__(128)
void head_kernel(const __bf16* __restrict__ h, const __bf16* __restrict__ W1t,
                 const float* __restrict__ b1, const float* __restrict__ w2,
                 const float* __restrict__ b2, float* __restrict__ out) {
  int wave = threadIdx.x >> 5;
  int lane = threadIdx.x & 31;
  int lr = lane & 15, hi = lane >> 4;
  long row0 = ((long)blockIdx.x * 4 + wave) * 16;

  long aRow = (row0 + lr) * 64;
  bf16x16 a0 = load_frag(h, aRow, 0,  hi);
  bf16x16 a1 = load_frag(h, aRow, 32, hi);

  float p[8];
#pragma unroll
  for (int r = 0; r < 8; ++r) p[r] = 0.0f;

#pragma unroll
  for (int t = 0; t < 8; ++t) {            // 8 N-tiles of fc1 (128 cols)
    long bRow = (long)(t * 16 + lr) * 64;
    bf16x16 bA = load_frag(W1t, bRow, 0,  hi);
    bf16x16 bB = load_frag(W1t, bRow, 32, hi);
    f32x8 acc;
#pragma unroll
    for (int r = 0; r < 8; ++r) acc[r] = 0.0f;
    acc = WMMA_BF16(a0, bA, acc);
    acc = WMMA_BF16(a1, bB, acc);
    int n = t * 16 + lr;
    float bw1 = b1[n], bw2 = w2[n];
#pragma unroll
    for (int r = 0; r < 8; ++r) {
      float v = acc[r] + bw1;
      v = fmaxf(v, 0.0f);
      p[r] += v * bw2;
    }
  }
#pragma unroll
  for (int r = 0; r < 8; ++r) {
    p[r] += __shfl_down(p[r], 8, 16);
    p[r] += __shfl_down(p[r], 4, 16);
    p[r] += __shfl_down(p[r], 2, 16);
    p[r] += __shfl_down(p[r], 1, 16);
  }
  if (lr == 0) {
    int mh = hi << 3;
#pragma unroll
    for (int r = 0; r < 8; ++r) out[row0 + mh + r] = p[r] + b2[0];
  }
}

// ------------------------------- host side ----------------------------------

extern "C" void kernel_launch(void* const* d_in, const int* in_sizes, int n_in,
                              void* d_out, int out_size, void* d_ws, size_t ws_size,
                              hipStream_t stream) {
  const float* x     = (const float*)d_in[0];
  const float* fc0_w = (const float*)d_in[1];
  const float* fc0_b = (const float*)d_in[2];
  const float* sw1   = (const float*)d_in[3];
  const float* sw2   = (const float*)d_in[4];
  const float* cw    = (const float*)d_in[5];
  /* cb (d_in[6]) cancels exactly under mean-subtracting BN -> unused */
  const float* bng   = (const float*)d_in[7];
  const float* bnb   = (const float*)d_in[8];
  const float* fc1_w = (const float*)d_in[9];
  const float* fc1_b = (const float*)d_in[10];
  const float* fc2_w = (const float*)d_in[11];
  const float* fc2_b = (const float*)d_in[12];
  float* out = (float*)d_out;

  // ---- carve workspace ----
  char* w = (char*)d_ws;
  size_t off = 0;
  auto alloc = [&](size_t bytes) -> void* {
    void* p = w + off;
    off = (off + bytes + 255) & ~(size_t)255;
    return p;
  };
  __bf16* hB   = (__bf16*)alloc(NROWS * 64 * 2);        // activation [bh][w][c]
  __bf16* hBt  = (__bf16*)alloc(NROWS * 64 * 2);        // activation [bh][c][w]
  float*  spec = (float*) alloc(NROWS * 64 * 4);        // layer output accum
  float*  Xf   = (float*) alloc(8L*256*48*64 * 4);      // fwd-W out (48 padded rows)
  __bf16* Xbt  = (__bf16*)alloc(8L*40*64*256 * 2);      // fwd-H Bt
  float*  Yf   = (float*) alloc(16L*20*80*64 * 4);      // fwd-H out (16 padded b)
  __bf16* Yb   = (__bf16*)alloc(16L*20*80*64 * 2);
  float*  Of   = (float*) alloc(16L*20*80*64 * 4);      // spectral out (16 padded b)
  __bf16* Obt  = (__bf16*)alloc(8L*20*64*96 * 2);       // inv-H Bt (96 padded K)
  float*  Zf   = (float*) alloc(8L*256*40*64 * 4);      // inv-H out
  __bf16* Zbt  = (__bf16*)alloc(8L*256*64*64 * 2);      // inv-W Bt (64 padded K)
  __bf16* DW   = (__bf16*)alloc(48L*256 * 2);
  __bf16* MH1  = (__bf16*)alloc(80L*256 * 2);
  __bf16* MH2  = (__bf16*)alloc(80L*256 * 2);
  __bf16* EH1  = (__bf16*)alloc(256L*96 * 2);
  __bf16* EH2  = (__bf16*)alloc(256L*96 * 2);
  __bf16* Gt   = (__bf16*)alloc(256L*64 * 2);
  __bf16* cwB  = (__bf16*)alloc(64L*64 * 2);
  __bf16* Wm1  = (__bf16*)alloc(40L*20*128*64 * 2);
  __bf16* Wm2  = (__bf16*)alloc(40L*20*128*64 * 2);
  __bf16* W1t  = (__bf16*)alloc(128L*64 * 2);
  float*  stats= (float*) alloc(128 * 4);

  // ---- one-time tables / weight casts ----
  build_dft_tables<<<96, 256, 0, stream>>>(DW, MH1, MH2, EH1, EH2, Gt);
  prep_w1t<<<32, 256, 0, stream>>>(fc1_w, W1t);

  // ---- lift ----
  fc0_kernel<<<(unsigned)((NROWS * 64 + 255) / 256), 256, 0, stream>>>(x, fc0_w, fc0_b, hB);
  transpose_hbt<<<8192, 256, 0, stream>>>(hB, hBt);

  const long sHb = 4194304, sHh = 16384;   // hB / hBt / spec strides per (b),(h)
  const long sXb = 786432,  sXh = 3072;    // Xf (48 rows per slab)
  const long sXtb = 655360;                // Xbt per-b stride (40*64*256)
  const long sYb = 102400,  sYk = 5120;    // Yf / Of strides

  for (int l = 0; l < 4; ++l) {
    prep_cwb<<<16, 256, 0, stream>>>(cw, cwB, l);
    prep_wm<<<(unsigned)((40L*20*128*64 + 255) / 256), 256, 0, stream>>>(sw1, sw2, Wm1, Wm2, l);

    // forward W DFT: X = DW @ h^T             [48x64 x K256] x (b,h)
    {
      dim3 g((48 / 16) * (64 / 64), 256, 8);
      gemm_wmma<<<g, 32, 0, stream>>>(DW, hBt, Xf, 48, 64, 256, 256, 256, 64,
                                      0, 0, 0, 0, 0, sHb, sHh, sXb, sXh);
    }
    cvt_xbt<<<8 * 40 * 4, 256, 0, stream>>>(Xf, Xbt);

    // forward H DFT, complex fused: Y = MH1@Xr + MH2@Xi  [80x64] x (b,kw)
    {
      dim3 g((80 / 16) * (64 / 64), 20, 8);
      gemm_wmma_dual<<<g, 32, 0, stream>>>(MH1, Xbt, MH2, Xbt, Yf,
                                           80, 64, 256, 256,
                                           256, 256, 256, 256, 64,
                                           0, 0, 0, 327680, 0,
                                           0, 0, sXtb, 16384,
                                           0, 0, sXtb, 16384,
                                           sYb, sYk, 0, 0, nullptr);
    }
    f32_to_bf16<<<(unsigned)((8L*20*80*64 + 255) / 256), 256, 0, stream>>>(Yf, Yb, 8L*20*80*64);

    // spectral channel mix, complex fused: [16x128 x K64+K64] x (j,kw)
    {
      dim3 g((16 / 16) * (128 / 64), 20, 40);
      gemm_wmma_dual<<<g, 32, 0, stream>>>(Yb, Wm1, Yb, Wm2, Of,
                                           16, 128, 64, 64,
                                           sYb, 64, sYb, 64, sYb,
                                           0, 0, 2560, 0, 0,
                                           64, sYk, 163840, 8192,
                                           64, sYk, 163840, 8192,
                                           64, sYk, 64, 2496, nullptr);
    }
    cvt_obt<<<(unsigned)((8L*20*64*96 + 255) / 256), 256, 0, stream>>>(Of, Obt);

    // inverse H: Zr / Zi                      [256x64 x K96] x (b,kw)
    {
      dim3 g((256 / 16) * (64 / 64), 20, 8);
      gemm_wmma<<<g, 32, 0, stream>>>(EH1, Obt, Zf, 256, 64, 96, 96, 96, 2560,
                                      0, 0, 0,    0, 0, 122880, 6144, 655360, 64);
      gemm_wmma<<<g, 32, 0, stream>>>(EH2, Obt, Zf, 256, 64, 96, 96, 96, 2560,
                                      0, 0, 1280, 0, 0, 122880, 6144, 655360, 64);
    }
    cvt_zbt<<<(unsigned)((8L*256*64*64 + 255) / 256), 256, 0, stream>>>(Zf, Zbt);

    // fused pointwise conv + inverse W + BN stats epilogue:
    //   spec = h @ cw^T + G @ Z ; stats += per-channel {sum, sumsq}
    hipMemsetAsync(stats, 0, 128 * sizeof(float), stream);
    {
      dim3 g((256 / 16) * (64 / 64), 256, 8);
      gemm_wmma_dual<<<g, 32, 0, stream>>>(hB, cwB, Gt, Zbt, spec,
                                           256, 64, 64, 64,
                                           64, 64, 64, 64, 64,
                                           0, 0, 0, 0, 0,
                                           sHb, sHh, 0, 0,
                                           0, 0, 1048576, 4096,
                                           sHb, sHh, 0, 0, stats);
    }

    // BatchNorm (+ReLU except last layer) -> next activation (both layouts)
    bn_apply<<<8192, 256, 0, stream>>>(
        spec, stats, bng + l * 64, bnb + l * 64, hB, hBt, (l < 3) ? 1 : 0);
  }

  // fused fc1 + relu + fc2 head (WMMA), 4 waves/block, 16 rows/wave
  head_kernel<<<(unsigned)(NROWS / 64), 128, 0, stream>>>(hB, W1t, fc1_b, fc2_w, fc2_b, out);

  (void)in_sizes; (void)n_in; (void)out_size; (void)ws_size;
}
